// CGGAT_36455682409139
// MI455X (gfx1250) — matrix operand order repaired and checked by reading, not compiled, not run on hardware
//
#include <hip/hip_runtime.h>
#include <math.h>

// ---------------------------------------------------------------------------
// CDNA5 (gfx1250) ALIGNN-style GNN forward.
// All dense GEMMs run on V_WMMA_F32_16X16X32_BF16 (bf16 A/B, f32 accum).
// Master activations kept in f32 (residual/LN/scatter math), bf16 shadow
// copies feed the WMMA GEMMs. Weights converted once per launch to bf16,
// transposed, K-padded to a multiple of 32.
// ---------------------------------------------------------------------------

typedef __attribute__((ext_vector_type(16))) __bf16 v16bf;
typedef __attribute__((ext_vector_type(8)))  __bf16 v8bf;
typedef __attribute__((ext_vector_type(8)))  float  v8f;

#define HID 128

__device__ __forceinline__ __bf16 f2bf(float f) {
  unsigned u = __float_as_uint(f);
  u += 0x7FFFu + ((u >> 16) & 1u);            // round-to-nearest-even
  unsigned short s = (unsigned short)(u >> 16);
  __bf16 r;
  __builtin_memcpy(&r, &s, sizeof(r));
  return r;
}

// --------------------------- weight / input prep ---------------------------

// Wt[n*Kpad + k] = bf16(W[k*N + n]), zero K-padding.
__global__ void k_wt_bf16(const float* __restrict__ W, __bf16* __restrict__ Wt,
                          int K, int Nn, int Kpad) {
  long idx = (long)blockIdx.x * blockDim.x + threadIdx.x;
  if (idx >= (long)Nn * Kpad) return;
  int n = (int)(idx / Kpad), k = (int)(idx % Kpad);
  float v = (k < K) ? W[(size_t)k * Nn + n] : 0.f;
  Wt[idx] = f2bf(v);
}

__global__ void k_pad_bf16(const float* __restrict__ X, __bf16* __restrict__ Y,
                           int rows, int cols, int colsPad) {
  long idx = (long)blockIdx.x * blockDim.x + threadIdx.x;
  if (idx >= (long)rows * colsPad) return;
  int r = (int)(idx / colsPad), c = (int)(idx % colsPad);
  float v = (c < cols) ? X[(size_t)r * cols + c] : 0.f;
  Y[idx] = f2bf(v);
}

// bondlength RBF: 80 bins over [0,8], output padded to 96 bf16 columns
__global__ void k_rbf_edge(const float* __restrict__ r, __bf16* __restrict__ out, int E) {
  int e = blockIdx.x * blockDim.x + threadIdx.x;
  if (e >= E) return;
  float x0 = r[3 * e], x1 = r[3 * e + 1], x2 = r[3 * e + 2];
  float d = sqrtf(x0 * x0 + x1 * x1 + x2 * x2);
  const float step = 8.0f / 79.0f, gamma = 1.0f / step;
  for (int i = 0; i < 96; ++i) {
    float v = 0.f;
    if (i < 80) { float df = d - (float)i * step; v = __expf(-gamma * df * df); }
    out[(size_t)e * 96 + i] = f2bf(v);
  }
}

// angle RBF: 40 bins over [-1,1], output padded to 64 bf16 columns
__global__ void k_rbf_angle(const float* __restrict__ h, __bf16* __restrict__ out, int T) {
  int t = blockIdx.x * blockDim.x + threadIdx.x;
  if (t >= T) return;
  float x = h[t];
  const float step = 2.0f / 39.0f, gamma = 1.0f / step;
  for (int i = 0; i < 64; ++i) {
    float v = 0.f;
    if (i < 40) { float df = x - (-1.0f + (float)i * step); v = __expf(-gamma * df * df); }
    out[(size_t)t * 64 + i] = f2bf(v);
  }
}

// ------------------------------ WMMA GEMM ----------------------------------
// One block = 16 rows x Nn cols. One wave per 16x16 tile (blockDim = 32*Nn/16).
// X: [M,Kpad] bf16 row-major. Wt: [Nn,Kpad] bf16 (weight transposed).
// mode 0: Yf = X@W (+bias)         (raw, f32 out)
// mode 1: LN(X@W+bias)*g+b -> SiLU (f32 out if Yf, bf16 shadow if Ybf)
__global__ __launch_bounds__(256) void k_gemm_bf16(
    const __bf16* __restrict__ X, const __bf16* __restrict__ Wt,
    const float* __restrict__ bias, const float* __restrict__ ln_g,
    const float* __restrict__ ln_b, float* __restrict__ Yf,
    __bf16* __restrict__ Ybf, int M, int Nn, int Kpad, int mode)
{
  extern __shared__ float smem[];
  const int wave = threadIdx.x >> 5;
  const int lane = threadIdx.x & 31;
  const int half = lane >> 4;     // K-half selector per ISA 16-bit striping
  const int l15  = lane & 15;
  const int m0 = blockIdx.x * 16;
  const int n0 = wave * 16;

  v8f acc = {};
  // A 16x32 bf16: lanes 0-15 row M=l15 K {0..7,16..23}; lanes 16-31 K {8..15,24..31}
  const __bf16* pA = X  + (size_t)(m0 + l15) * Kpad + half * 8;
  // B 32x16 bf16: lanes 0-15 col N=l15 K {0..7,16..23}; lanes 16-31 K {8..15,24..31}
  const __bf16* pB = Wt + (size_t)(n0 + l15) * Kpad + half * 8;

  for (int k0 = 0; k0 < Kpad; k0 += 32) {
    v8bf a_lo = *(const v8bf*)(pA + k0);
    v8bf a_hi = *(const v8bf*)(pA + k0 + 16);
    v8bf b_lo = *(const v8bf*)(pB + k0);
    v8bf b_hi = *(const v8bf*)(pB + k0 + 16);
    v16bf a, b;
#pragma unroll
    for (int i = 0; i < 8; ++i) {
      a[i] = a_lo[i]; a[i + 8] = a_hi[i];
      b[i] = b_lo[i]; b[i + 8] = b_hi[i];
    }
    acc = __builtin_amdgcn_wmma_f32_16x16x32_bf16(false, a, false, b,
                                                  (short)0, acc, false, false);
  }

  const float bs = bias ? bias[n0 + l15] : 0.f;

  if (mode == 0) {
    // C/D layout: VGPR j holds row (half*8 + j), col l15
#pragma unroll
    for (int j = 0; j < 8; ++j) {
      int mr = m0 + half * 8 + j;
      Yf[(size_t)mr * Nn + n0 + l15] = acc[j] + bs;
    }
    return;
  }

  // mode 1: fused bias + LayerNorm + SiLU across the full Nn row in LDS
  float* rowbuf = smem;                 // [16][Nn]
  float* meanb  = smem + 16 * Nn;       // [16]
  float* rstdb  = meanb + 16;           // [16]
#pragma unroll
  for (int j = 0; j < 8; ++j)
    rowbuf[(half * 8 + j) * Nn + n0 + l15] = acc[j] + bs;
  __syncthreads();
  int tid = threadIdx.x;
  if (tid < 16) {
    float s = 0.f;
    for (int c = 0; c < Nn; ++c) s += rowbuf[tid * Nn + c];
    float mu = s / (float)Nn;
    float v2 = 0.f;
    for (int c = 0; c < Nn; ++c) { float d = rowbuf[tid * Nn + c] - mu; v2 += d * d; }
    meanb[tid] = mu;
    rstdb[tid] = rsqrtf(v2 / (float)Nn + 1e-5f);
  }
  __syncthreads();
  for (int idx = tid; idx < 16 * Nn; idx += blockDim.x) {
    int mr = idx / Nn, c = idx - mr * Nn;
    float t  = (rowbuf[idx] - meanb[mr]) * rstdb[mr] * ln_g[c] + ln_b[c];
    float si = t / (1.f + __expf(-t));
    size_t o = (size_t)(m0 + mr) * Nn + c;
    if (Yf)  Yf[o]  = si;
    if (Ybf) Ybf[o] = f2bf(si);
  }
}

// --------------------- gated conv gather / scatter -------------------------
// z holds e@Weg+bg on entry; becomes full gate pre-activation on exit.
__global__ void k_gather_gate(float* __restrict__ z, const float* __restrict__ hs,
                              const float* __restrict__ hd, const float* __restrict__ hu,
                              const int* __restrict__ src, const int* __restrict__ dst,
                              float* __restrict__ num, float* __restrict__ den, long nE)
{
  long idx = (long)blockIdx.x * blockDim.x + threadIdx.x;
  if (idx >= nE * HID) return;
  long e = idx >> 7;
  int  c = (int)(idx & 127);
  int  s = src[e], d = dst[e];
  float zz  = z[idx] + hs[(size_t)s * HID + c] + hd[(size_t)d * HID + c];
  float sig = 1.f / (1.f + __expf(-zz));
  float m   = sig * hu[(size_t)s * HID + c];
  z[idx] = zz;
  atomicAdd(&num[(size_t)d * HID + c], m);
  atomicAdd(&den[(size_t)d * HID + c], sig);
}

// x += SiLU(LN(hdu + num/(den+eps))); one wave per 128-wide row (wave32)
__global__ void k_node_update(float* __restrict__ x, __bf16* __restrict__ xb,
                              const float* __restrict__ hdu, const float* __restrict__ num,
                              const float* __restrict__ den, const float* __restrict__ g,
                              const float* __restrict__ b, int n)
{
  int row  = blockIdx.x * (blockDim.x >> 5) + (threadIdx.x >> 5);
  int lane = threadIdx.x & 31;
  if (row >= n) return;
  size_t base = (size_t)row * HID;
  float vals[4];
  float s = 0.f;
#pragma unroll
  for (int i = 0; i < 4; ++i) {
    int c = lane * 4 + i;
    float agg = num[base + c] / (den[base + c] + 1e-6f);
    vals[i] = hdu[base + c] + agg;
    s += vals[i];
  }
#pragma unroll
  for (int off = 16; off > 0; off >>= 1) s += __shfl_xor(s, off, 32);
  float mu = s * (1.f / HID);
  float v2 = 0.f;
#pragma unroll
  for (int i = 0; i < 4; ++i) { float d = vals[i] - mu; v2 += d * d; }
#pragma unroll
  for (int off = 16; off > 0; off >>= 1) v2 += __shfl_xor(v2, off, 32);
  float rstd = rsqrtf(v2 * (1.f / HID) + 1e-5f);
#pragma unroll
  for (int i = 0; i < 4; ++i) {
    int c = lane * 4 + i;
    float t  = (vals[i] - mu) * rstd * g[c] + b[c];
    float si = t / (1.f + __expf(-t));
    float xn = x[base + c] + si;
    x[base + c]  = xn;
    xb[base + c] = f2bf(xn);
  }
}

// e += SiLU(LN(z)); same structure
__global__ void k_edge_update(float* __restrict__ e, __bf16* __restrict__ eb,
                              const float* __restrict__ z, const float* __restrict__ g,
                              const float* __restrict__ b, int n)
{
  int row  = blockIdx.x * (blockDim.x >> 5) + (threadIdx.x >> 5);
  int lane = threadIdx.x & 31;
  if (row >= n) return;
  size_t base = (size_t)row * HID;
  float vals[4];
  float s = 0.f;
#pragma unroll
  for (int i = 0; i < 4; ++i) { vals[i] = z[base + lane * 4 + i]; s += vals[i]; }
#pragma unroll
  for (int off = 16; off > 0; off >>= 1) s += __shfl_xor(s, off, 32);
  float mu = s * (1.f / HID);
  float v2 = 0.f;
#pragma unroll
  for (int i = 0; i < 4; ++i) { float d = vals[i] - mu; v2 += d * d; }
#pragma unroll
  for (int off = 16; off > 0; off >>= 1) v2 += __shfl_xor(v2, off, 32);
  float rstd = rsqrtf(v2 * (1.f / HID) + 1e-5f);
#pragma unroll
  for (int i = 0; i < 4; ++i) {
    int c = lane * 4 + i;
    float t  = (vals[i] - mu) * rstd * g[c] + b[c];
    float si = t / (1.f + __expf(-t));
    float en = e[base + c] + si;
    e[base + c]  = en;
    eb[base + c] = f2bf(en);
  }
}

// ------------------------------ pooling / head -----------------------------
__global__ void k_pool(const float* __restrict__ v, const int* __restrict__ n2g,
                       float* __restrict__ sums, float* __restrict__ counts, long n)
{
  long idx = (long)blockIdx.x * blockDim.x + threadIdx.x;
  if (idx >= n * HID) return;
  long row = idx >> 7;
  int  c   = (int)(idx & 127);
  int  g   = n2g[row];
  atomicAdd(&sums[(size_t)g * HID + c], v[idx]);
  if (c == 0) atomicAdd(&counts[g], 1.f);
}

__global__ void k_fc(const float* __restrict__ sums, const float* __restrict__ counts,
                     const float* __restrict__ W, const float* __restrict__ b,
                     float* __restrict__ out)
{
  int g = blockIdx.x, lane = threadIdx.x;
  float cnt = fmaxf(counts[g], 1.f);
  float acc = 0.f;
  for (int c = lane; c < HID; c += 32) acc += (sums[(size_t)g * HID + c] / cnt) * W[c];
#pragma unroll
  for (int off = 16; off > 0; off >>= 1) acc += __shfl_xor(acc, off, 32);
  if (lane == 0) out[g] = acc + b[0];
}

// ------------------------------- host side ---------------------------------
// Input flattening assumption (jax pytree: dict keys sorted, lists in order):
//  params:
//   ang1  {W,b,be,g}            -> 0..3
//   ang2                        -> 4..7
//   atom                        -> 8..11
//   edge1                       -> 12..15
//   edge2                       -> 16..19
//   edge_convs[0..2] each {Wdg,Wdu,Weg,Wsg,Wsu,bdu,bg,bn1,bn2,bsu,gn1,gn2}
//                               -> 20..31, 32..43, 44..55
//   fc {W,b}                    -> 56,57
//   final (conv dict)           -> 58..69
//   node_convs[0..2]            -> 70..81, 82..93, 94..105
//  atom_features=106, r=107, angle_h=108, src=109, dst=110,
//  esrc=111, edst=112, node2graph=113

extern "C" void kernel_launch(void* const* d_in, const int* in_sizes, int n_in,
                              void* d_out, int out_size, void* d_ws, size_t ws_size,
                              hipStream_t stream) {
  (void)n_in; (void)out_size; (void)ws_size;
  const int N = in_sizes[106] / 92;
  const int E = in_sizes[107] / 3;
  const int T = in_sizes[108];

  auto F = [&](int i) { return (const float*)d_in[i]; };
  auto I = [&](int i) { return (const int*)d_in[i]; };

  // ---- workspace bump allocator (256B aligned) ----
  char* wp = (char*)d_ws;
  auto alloc = [&](size_t bytes) -> void* {
    void* r = wp; wp += (bytes + 255) & ~(size_t)255; return r;
  };
  float*  v_f = (float*) alloc((size_t)N * HID * 4);
  __bf16* v_b = (__bf16*)alloc((size_t)N * HID * 2);
  float*  e_f = (float*) alloc((size_t)E * HID * 4);
  __bf16* e_b = (__bf16*)alloc((size_t)E * HID * 2);
  float*  a_f = (float*) alloc((size_t)T * HID * 4);
  __bf16* a_b = (__bf16*)alloc((size_t)T * HID * 2);
  float* hs  = (float*)alloc((size_t)E * HID * 4);   // conv "node"-side GEMMs,
  float* hd  = (float*)alloc((size_t)E * HID * 4);   // sized for line graph (E)
  float* hu  = (float*)alloc((size_t)E * HID * 4);
  float* hdu = (float*)alloc((size_t)E * HID * 4);
  float* z   = (float*)alloc((size_t)T * HID * 4);   // conv "edge"-side (T)
  float* num = (float*)alloc((size_t)E * HID * 4);
  float* den = (float*)alloc((size_t)E * HID * 4);
  __bf16* rbf_buf = (__bf16*)alloc((size_t)T * 64 * 2);  // also atom pad / edge rbf
  __bf16* emb_buf = (__bf16*)alloc((size_t)T * 64 * 2);
  float* sums   = (float*)alloc(32 * HID * 4);
  float* counts = (float*)alloc(32 * 4);
  __bf16* atomWt = (__bf16*)alloc(128 * 96 * 2);
  __bf16* e1Wt   = (__bf16*)alloc(64  * 96 * 2);
  __bf16* e2Wt   = (__bf16*)alloc(128 * 64 * 2);
  __bf16* a1Wt   = (__bf16*)alloc(64  * 64 * 2);
  __bf16* a2Wt   = (__bf16*)alloc(128 * 64 * 2);
  __bf16* convWt = (__bf16*)alloc((size_t)7 * 5 * 128 * 128 * 2);

  auto cvt_wt = [&](const float* W, __bf16* Wt, int K, int Nn, int Kpad) {
    long tot = (long)Nn * Kpad;
    k_wt_bf16<<<(unsigned)((tot + 255) / 256), 256, 0, stream>>>(W, Wt, K, Nn, Kpad);
  };
  cvt_wt(F(8),  atomWt, 92, 128, 96);   // atom.W
  cvt_wt(F(12), e1Wt,   80, 64,  96);   // edge1.W
  cvt_wt(F(16), e2Wt,   64, 128, 64);   // edge2.W
  cvt_wt(F(0),  a1Wt,   40, 64,  64);   // ang1.W
  cvt_wt(F(4),  a2Wt,   64, 128, 64);   // ang2.W

  // conv execution order: node0, edge0, node1, edge1, node2, edge2, final
  const int cbase[7] = {70, 20, 82, 32, 94, 44, 58};
  const int wofs[5]  = {3, 0, 2, 4, 1};  // Wsg, Wdg, Weg, Wsu, Wdu (sorted-key offsets)
  for (int c = 0; c < 7; ++c)
    for (int w = 0; w < 5; ++w)
      cvt_wt(F(cbase[c] + wofs[w]), convWt + ((size_t)c * 5 + w) * 16384, 128, 128, 128);

  auto gemm = [&](const __bf16* X, const __bf16* Wt, const float* bias,
                  const float* g, const float* b, float* Yf, __bf16* Ybf,
                  int M, int Nn, int Kpad, int mode) {
    dim3 grid((unsigned)(M / 16));
    dim3 block((unsigned)(32 * (Nn / 16)));
    size_t sh = (mode == 1) ? (size_t)(16 * Nn + 32) * 4 : 0;
    k_gemm_bf16<<<grid, block, sh, stream>>>(X, Wt, bias, g, b, Yf, Ybf, M, Nn, Kpad, mode);
  };

  // ---- embeddings ----
  {
    long tot = (long)N * 96;
    k_pad_bf16<<<(unsigned)((tot + 255) / 256), 256, 0, stream>>>(F(106), rbf_buf, N, 92, 96);
    gemm(rbf_buf, atomWt, F(9), F(11), F(10), v_f, v_b, N, 128, 96, 1);   // v

    k_rbf_edge<<<(unsigned)((E + 255) / 256), 256, 0, stream>>>(F(107), rbf_buf, E);
    gemm(rbf_buf, e1Wt, F(13), F(15), F(14), nullptr, emb_buf, E, 64, 96, 1);
    gemm(emb_buf, e2Wt, F(17), F(19), F(18), e_f, e_b, E, 128, 64, 1);    // e

    k_rbf_angle<<<(unsigned)((T + 255) / 256), 256, 0, stream>>>(F(108), rbf_buf, T);
    gemm(rbf_buf, a1Wt, F(1), F(3), F(2), nullptr, emb_buf, T, 64, 64, 1);
    gemm(emb_buf, a2Wt, F(5), F(7), F(6), a_f, a_b, T, 128, 64, 1);       // a
  }

  auto run_conv = [&](int pb, float* xf, __bf16* xb, float* ef, __bf16* eb,
                      const int* csrc, const int* cdst, int nN, int nE, __bf16* wts) {
    __bf16* Wsg = wts;
    __bf16* Wdg = wts + 1 * 16384;
    __bf16* Weg = wts + 2 * 16384;
    __bf16* Wsu = wts + 3 * 16384;
    __bf16* Wdu = wts + 4 * 16384;
    const float* bdu = F(pb + 5);
    const float* bg  = F(pb + 6);
    const float* bn1 = F(pb + 7);
    const float* bn2 = F(pb + 8);
    const float* bsu = F(pb + 9);
    const float* gn1 = F(pb + 10);
    const float* gn2 = F(pb + 11);
    gemm(xb, Wsg, nullptr, nullptr, nullptr, hs,  nullptr, nN, 128, 128, 0);
    gemm(xb, Wdg, nullptr, nullptr, nullptr, hd,  nullptr, nN, 128, 128, 0);
    gemm(xb, Wsu, bsu,     nullptr, nullptr, hu,  nullptr, nN, 128, 128, 0);
    gemm(xb, Wdu, bdu,     nullptr, nullptr, hdu, nullptr, nN, 128, 128, 0);
    gemm(eb, Weg, bg,      nullptr, nullptr, z,   nullptr, nE, 128, 128, 0);
    hipMemsetAsync(num, 0, (size_t)nN * HID * 4, stream);
    hipMemsetAsync(den, 0, (size_t)nN * HID * 4, stream);
    long tot = (long)nE * HID;
    k_gather_gate<<<(unsigned)((tot + 255) / 256), 256, 0, stream>>>(
        z, hs, hd, hu, csrc, cdst, num, den, (long)nE);
    k_node_update<<<(unsigned)((nN + 7) / 8), 256, 0, stream>>>(
        xf, xb, hdu, num, den, gn1, bn1, nN);
    k_edge_update<<<(unsigned)((nE + 7) / 8), 256, 0, stream>>>(
        ef, eb, z, gn2, bn2, nE);
  };

  const int* src  = I(109);
  const int* dst  = I(110);
  const int* esrc = I(111);
  const int* edst = I(112);

  for (int c = 0; c < 7; ++c) {
    __bf16* wts = convWt + (size_t)c * 5 * 16384;
    if ((c & 1) == 0)  // node-graph convs (incl. final at c==6)
      run_conv(cbase[c], v_f, v_b, e_f, e_b, src, dst, N, E, wts);
    else               // line-graph convs
      run_conv(cbase[c], e_f, e_b, a_f, a_b, esrc, edst, E, T, wts);
  }

  // ---- avg pooling + head ----
  hipMemsetAsync(sums,   0, 32 * HID * 4, stream);
  hipMemsetAsync(counts, 0, 32 * 4,       stream);
  long totp = (long)N * HID;
  k_pool<<<(unsigned)((totp + 255) / 256), 256, 0, stream>>>(v_f, I(113), sums, counts, (long)N);
  k_fc<<<32, 32, 0, stream>>>(sums, counts, F(56), F(57), (float*)d_out);
}
